// DITAttention_78950088835077
// MI455X (gfx1250) — matrix operand order repaired and checked by reading, not compiled, not run on hardware
//
#include <hip/hip_runtime.h>
#include <hip/hip_bf16.h>
#include <math.h>

// Problem constants (from reference): B=4, S=2048, D=1024, H=16, HD=64
#define BB 4
#define SS 2048
#define DD 1024
#define HH 16
#define HD 64

typedef _Float16 v16h __attribute__((ext_vector_type(16)));
typedef _Float16 v8h  __attribute__((ext_vector_type(8)));
typedef _Float16 v4h  __attribute__((ext_vector_type(4)));
typedef float    v8f  __attribute__((ext_vector_type(8)));

__device__ inline v8f zero8() {
    v8f z = {0.f, 0.f, 0.f, 0.f, 0.f, 0.f, 0.f, 0.f};
    return z;
}

__device__ inline v8f wmma_f16(v16h a, v16h b, v8f c) {
    // D = A(16x32 f16) * B(32x16 f16) + C(16x16 f32)
    return __builtin_amdgcn_wmma_f32_16x16x32_f16(
        /*neg_a=*/false, a, /*neg_b=*/false, b,
        /*c_mod=*/(short)0, c, /*reuse_a=*/false, /*reuse_b=*/false);
}

// ---- Fragment loaders (contiguous K = half*16 + s in both A and B) ----

// A-matrix 16x32 f16: lane -> M = lane&15, half = lane>>4.
// K pattern per lane = {half*16 .. half*16+15} (contiguous, matches B):
// two contiguous 16-byte chunks -> two v8h loads.
__device__ inline v16h load_a_f16(const _Float16* __restrict__ base, int ld, int lane) {
    const int m = lane & 15, half = lane >> 4;
    const _Float16* row = base + (long)m * ld + half * 16;
    const v8h lo = *(const v8h*)row;
    const v8h hi = *(const v8h*)(row + 8);
    return __builtin_shufflevector(lo, hi, 0, 1, 2, 3, 4, 5, 6, 7,
                                   8, 9, 10, 11, 12, 13, 14, 15);
}

// B-matrix 32x16: lane -> N = lane&15; K = half*16 + v*2 + s: per lane 16
// contiguous K elements. `p` points at (row=k0+half*16, col=n), K-contiguous.
__device__ inline v16h load_b_f16_contig(const _Float16* __restrict__ p) {
    const v8h lo = *(const v8h*)p;
    const v8h hi = *(const v8h*)(p + 8);
    return __builtin_shufflevector(lo, hi, 0, 1, 2, 3, 4, 5, 6, 7,
                                   8, 9, 10, 11, 12, 13, 14, 15);
}

// B fragment for C = X * Wh^T: B[k][n] = Wh[(e0+n)*ld + kk + k] (f16).
__device__ inline v16h load_b_wT_f16(const _Float16* __restrict__ W, int ld, int e0,
                                     int kk, int lane) {
    const int n = lane & 15, half = lane >> 4;
    return load_b_f16_contig(W + (long)(e0 + n) * ld + kk + half * 16);
}

// =====================================================================
// Kernel 0: one-shot f32 -> f16 conversion (X and the four weights).
// =====================================================================
__global__ __launch_bounds__(256) void cvt_f16_kernel(const float* __restrict__ src,
                                                      _Float16* __restrict__ dst,
                                                      int n4) {
    const int i = (blockIdx.x * 256 + threadIdx.x);
    if (i < n4) {
        const float4 v = ((const float4*)src)[i];
        v4h o = {(_Float16)v.x, (_Float16)v.y, (_Float16)v.z, (_Float16)v.w};
        ((v4h*)dst)[i] = o;
    }
}

// =====================================================================
// Kernel 1: fused Q/K/V projection (Xh * Wh^T) + RoPE epilogue.
// Wave -> one 16-row M tile x one (dim, dim+32) tile pair inside one head.
// Writes Qh,Kh as [b,h,s,d] f16 and V transposed Vt as [b,h,d,s] f16.
// =====================================================================
__global__ __launch_bounds__(256) void qkv_rope_kernel(
    const _Float16* __restrict__ Xh, const _Float16* __restrict__ Wq,
    const _Float16* __restrict__ Wk, const _Float16* __restrict__ Wv,
    _Float16* __restrict__ Qh, _Float16* __restrict__ Kh,
    _Float16* __restrict__ Vt) {
    const int lane = threadIdx.x & 31;
    const int wid = blockIdx.x * (blockDim.x >> 5) + (threadIdx.x >> 5);
    const int mTile = wid >> 5;     // 0..511
    const int pairI = wid & 31;     // head*2 + t
    const int h = pairI >> 1;
    const int t = pairI & 1;
    const int g0 = mTile * 16;      // global token row
    const int b = g0 >> 11;         // / 2048
    const int s0 = g0 & (SS - 1);
    const int e1 = h * HD + t * 16; // columns e1..e1+15 pair with e1+32..
    const int e2 = e1 + 32;

    v8f qa = zero8(), qb = zero8();
    v8f ka = zero8(), kb = zero8();
    v8f va = zero8(), vb = zero8();

    const _Float16* xrow = Xh + (long)g0 * DD;
    for (int kk = 0; kk < DD; kk += 32) {
        __builtin_prefetch(xrow + kk + 128, 0, 1);
        v16h a = load_a_f16(xrow + kk, DD, lane);
        v16h bq1 = load_b_wT_f16(Wq, DD, e1, kk, lane);
        v16h bq2 = load_b_wT_f16(Wq, DD, e2, kk, lane);
        v16h bk1 = load_b_wT_f16(Wk, DD, e1, kk, lane);
        v16h bk2 = load_b_wT_f16(Wk, DD, e2, kk, lane);
        v16h bv1 = load_b_wT_f16(Wv, DD, e1, kk, lane);
        v16h bv2 = load_b_wT_f16(Wv, DD, e2, kk, lane);
        qa = wmma_f16(a, bq1, qa);
        qb = wmma_f16(a, bq2, qb);
        ka = wmma_f16(a, bk1, ka);
        kb = wmma_f16(a, bk2, kb);
        va = wmma_f16(a, bv1, va);
        vb = wmma_f16(a, bv2, vb);
    }

    // Epilogue: RoPE rotate (d, d+32) pairs of Q and K; V passthrough.
    const int n = lane & 15, half = lane >> 4;
    const int d1 = t * 16 + n;           // dim within head, first half
    const int d2 = d1 + 32;
    // inv_freq[i] = 10000^(-i/32) = exp2(-i/32 * log2(10000))
    const float invf = exp2f(-((float)d1 / 32.0f) * 13.287712379549449f);
    const long bh = (long)(b * HH + h);
#pragma unroll
    for (int r = 0; r < 8; ++r) {
        const int m = r + half * 8;
        const int s = s0 + m;
        const float ang = (float)s * invf;
        const float c = cosf(ang), sn = sinf(ang);
        const float q1 = qa[r] * c - qb[r] * sn;
        const float q2 = qa[r] * sn + qb[r] * c;
        const float k1 = ka[r] * c - kb[r] * sn;
        const float k2 = ka[r] * sn + kb[r] * c;
        const long roff = (bh * SS + s) * HD;
        Qh[roff + d1] = (_Float16)q1;
        Qh[roff + d2] = (_Float16)q2;
        Kh[roff + d1] = (_Float16)k1;
        Kh[roff + d2] = (_Float16)k2;
        Vt[(bh * HD + d1) * SS + s] = (_Float16)va[r];
        Vt[(bh * HD + d2) * SS + s] = (_Float16)vb[r];
    }
}

// =====================================================================
// Kernel 2: flash attention. One wave = 16 query rows of one (b,h).
// Streams 32 keys/iter: 4 WMMA for QK^T, online softmax (xor-shuffle row
// reductions), P -> f16 via per-wave LDS transpose, 4 WMMA for P*V.
// =====================================================================
__global__ __launch_bounds__(256) void flash_attn_kernel(
    const _Float16* __restrict__ Qh, const _Float16* __restrict__ Kh,
    const _Float16* __restrict__ Vt, const int* __restrict__ mask,
    _Float16* __restrict__ Aout) {
    __shared__ __align__(16) _Float16 ldsP[8][16 * 32];  // per-wave 16x32 P tile

    const int lane = threadIdx.x & 31;
    const int wslot = threadIdx.x >> 5;
    const int wid = blockIdx.x * 8 + wslot;
    const int bh = wid >> 7;               // (b*H + h)
    const int q0 = (wid & 127) << 4;       // query tile start
    const int b = bh >> 4;
    const int h = bh & 15;
    const int n = lane & 15, half = lane >> 4;

    const _Float16* qp = Qh + ((long)bh * SS + q0) * HD;
    const v16h qa0 = load_a_f16(qp, HD, lane);       // dims 0..31
    const v16h qa1 = load_a_f16(qp + 32, HD, lane);  // dims 32..63

    v8f o0 = zero8(), o1 = zero8(), o2 = zero8(), o3 = zero8();
    float mi[8], li[8];
#pragma unroll
    for (int r = 0; r < 8; ++r) { mi[r] = -3.0e38f; li[r] = 0.0f; }

    const _Float16* kbase = Kh + (long)bh * SS * HD;
    const _Float16* vbase = Vt + (long)bh * HD * SS;
    const int* mrow = mask + b * SS;
    _Float16* lp = ldsP[wslot];

    for (int j = 0; j < SS; j += 32) {
        // ---- S = Q * K^T for 32 keys (two 16-key subtiles) ----
        v8f sA = zero8(), sB = zero8();
        {
            const _Float16* kp = kbase + (long)(j + n) * HD + half * 16;
            sA = wmma_f16(qa0, load_b_f16_contig(kp), sA);
            sA = wmma_f16(qa1, load_b_f16_contig(kp + 32), sA);
            const _Float16* kp2 = kp + 16 * HD;
            sB = wmma_f16(qa0, load_b_f16_contig(kp2), sB);
            sB = wmma_f16(qa1, load_b_f16_contig(kp2 + 32), sB);
        }
        const float add0 = mrow[j + n] ? 0.0f : -3.0e38f;
        const float add1 = mrow[j + 16 + n] ? 0.0f : -3.0e38f;

        // ---- online softmax: row max / exp / row sum ----
        float mt[8];
#pragma unroll
        for (int r = 0; r < 8; ++r) {
            sA[r] = sA[r] * 0.125f + add0;  // 1/sqrt(64)
            sB[r] = sB[r] * 0.125f + add1;
            float v = fmaxf(sA[r], sB[r]);
#pragma unroll
            for (int x = 1; x < 16; x <<= 1) v = fmaxf(v, __shfl_xor(v, x, 32));
            mt[r] = v;
        }
        float alpha[8];
#pragma unroll
        for (int r = 0; r < 8; ++r) {
            const float mn = fmaxf(mi[r], mt[r]);
            alpha[r] = expf(mi[r] - mn);
            mi[r] = mn;
        }
#pragma unroll
        for (int r = 0; r < 8; ++r) {
            const float p0 = expf(sA[r] - mi[r]);
            const float p1 = expf(sB[r] - mi[r]);
            sA[r] = p0;
            sB[r] = p1;
            float rs = p0 + p1;
#pragma unroll
            for (int x = 1; x < 16; x <<= 1) rs += __shfl_xor(rs, x, 32);
            li[r] = li[r] * alpha[r] + rs;
        }

        // ---- C-layout P -> LDS (row-major 16x32 f16) -> A-fragment ----
#pragma unroll
        for (int r = 0; r < 8; ++r) {
            const int m = r + half * 8;
            lp[m * 32 + n] = (_Float16)sA[r];
            lp[m * 32 + 16 + n] = (_Float16)sB[r];
        }
        asm volatile("s_wait_dscnt 0x0" ::: "memory");  // in-order LDS, cross-lane read
        const v16h pa = load_a_f16(lp, 32, lane);

        // ---- rescale O, then O += P * V ----
#pragma unroll
        for (int r = 0; r < 8; ++r) {
            o0[r] *= alpha[r]; o1[r] *= alpha[r];
            o2[r] *= alpha[r]; o3[r] *= alpha[r];
        }
        const _Float16* vp = vbase + (long)n * SS + j + half * 16;
        o0 = wmma_f16(pa, load_b_f16_contig(vp), o0);
        o1 = wmma_f16(pa, load_b_f16_contig(vp + 16 * SS), o1);
        o2 = wmma_f16(pa, load_b_f16_contig(vp + 32 * SS), o2);
        o3 = wmma_f16(pa, load_b_f16_contig(vp + 48 * SS), o3);
    }

    // ---- normalize and store attn-out as [b,s,h,d] f16 ----
#pragma unroll
    for (int r = 0; r < 8; ++r) {
        const int m = r + half * 8;
        const float inv_l = 1.0f / li[r];
        _Float16* op = Aout + ((long)b * SS + q0 + m) * DD + h * HD;
        op[n]      = (_Float16)(o0[r] * inv_l);
        op[16 + n] = (_Float16)(o1[r] * inv_l);
        op[32 + n] = (_Float16)(o2[r] * inv_l);
        op[48 + n] = (_Float16)(o3[r] * inv_l);
    }
}

// =====================================================================
// Kernel 3: output projection  out = Aout(f16) * Woh^T, f32 result.
// Each wave computes 4 N-tiles (64 columns) sharing the A fragment.
// =====================================================================
__global__ __launch_bounds__(256) void out_proj_kernel(
    const _Float16* __restrict__ Aout, const _Float16* __restrict__ Woh,
    float* __restrict__ Out) {
    const int lane = threadIdx.x & 31;
    const int wid = blockIdx.x * (blockDim.x >> 5) + (threadIdx.x >> 5);
    const int mTile = wid >> 4;       // 0..511
    const int nG = wid & 15;          // 0..15 -> 64 columns each
    const int g0 = mTile * 16;
    const int e0 = nG * 64;

    v8f a0 = zero8(), a1 = zero8(), a2 = zero8(), a3 = zero8();
    const _Float16* arow = Aout + (long)g0 * DD;
    for (int kk = 0; kk < DD; kk += 32) {
        __builtin_prefetch(arow + kk + 128, 0, 1);
        v16h a = load_a_f16(arow + kk, DD, lane);
        v16h b0 = load_b_wT_f16(Woh, DD, e0, kk, lane);
        v16h b1 = load_b_wT_f16(Woh, DD, e0 + 16, kk, lane);
        v16h b2 = load_b_wT_f16(Woh, DD, e0 + 32, kk, lane);
        v16h b3 = load_b_wT_f16(Woh, DD, e0 + 48, kk, lane);
        a0 = wmma_f16(a, b0, a0);
        a1 = wmma_f16(a, b1, a1);
        a2 = wmma_f16(a, b2, a2);
        a3 = wmma_f16(a, b3, a3);
    }
    const int n = lane & 15, half = lane >> 4;
#pragma unroll
    for (int r = 0; r < 8; ++r) {
        float* orow = Out + (long)(g0 + r + half * 8) * DD + e0 + n;
        orow[0]  = a0[r];
        orow[16] = a1[r];
        orow[32] = a2[r];
        orow[48] = a3[r];
    }
}

// =====================================================================
extern "C" void kernel_launch(void* const* d_in, const int* in_sizes, int n_in,
                              void* d_out, int out_size, void* d_ws, size_t ws_size,
                              hipStream_t stream) {
    (void)in_sizes; (void)n_in; (void)out_size; (void)ws_size;
    const float* X    = (const float*)d_in[0];
    const int*   mask = (const int*)d_in[1];
    const float* Wq   = (const float*)d_in[2];
    const float* Wk   = (const float*)d_in[3];
    const float* Wv   = (const float*)d_in[4];
    const float* Wo   = (const float*)d_in[5];
    float* out = (float*)d_out;

    char* ws = (char*)d_ws;
    const size_t XB = (size_t)BB * SS * DD * sizeof(_Float16);  // 16 MiB
    const size_t WB = (size_t)DD * DD * sizeof(_Float16);       // 2 MiB
    _Float16* Qh  = (_Float16*)(ws);
    _Float16* Kh  = (_Float16*)(ws + XB);
    _Float16* Vt  = (_Float16*)(ws + 2 * XB);
    _Float16* Ao  = (_Float16*)(ws + 3 * XB);
    _Float16* Xh  = (_Float16*)(ws + 4 * XB);
    _Float16* Wqh = (_Float16*)(ws + 5 * XB);
    _Float16* Wkh = (_Float16*)(ws + 5 * XB + WB);
    _Float16* Wvh = (_Float16*)(ws + 5 * XB + 2 * WB);
    _Float16* Woh = (_Float16*)(ws + 5 * XB + 3 * WB);

    // ---- one-shot f32 -> f16 conversions ----
    const int nX4 = BB * SS * DD / 4;   // 2,097,152
    const int nW4 = DD * DD / 4;        // 262,144
    cvt_f16_kernel<<<(nX4 + 255) / 256, 256, 0, stream>>>(X, Xh, nX4);
    cvt_f16_kernel<<<(nW4 + 255) / 256, 256, 0, stream>>>(Wq, Wqh, nW4);
    cvt_f16_kernel<<<(nW4 + 255) / 256, 256, 0, stream>>>(Wk, Wkh, nW4);
    cvt_f16_kernel<<<(nW4 + 255) / 256, 256, 0, stream>>>(Wv, Wvh, nW4);
    cvt_f16_kernel<<<(nW4 + 255) / 256, 256, 0, stream>>>(Wo, Woh, nW4);

    // 512 M-tiles * 32 (head,t) pairs = 16384 waves / 8 waves per block
    qkv_rope_kernel<<<2048, 256, 0, stream>>>(Xh, Wqh, Wkh, Wvh, Qh, Kh, Vt);
    // 64 (b,h) * 128 q-tiles = 8192 waves
    flash_attn_kernel<<<1024, 256, 0, stream>>>(Qh, Kh, Vt, mask, Ao);
    // 512 M-tiles * 16 N-groups (4 tiles each) = 8192 waves
    out_proj_kernel<<<1024, 256, 0, stream>>>(Ao, Woh, out);
}
